// CGCNNConv_89515708383412
// MI455X (gfx1250) — compile-verified, matrix operand order checked by reading, not run on hardware
//
#include <hip/hip_runtime.h>
#include <hip/hip_bf16.h>
#include <math.h>

typedef __attribute__((ext_vector_type(16))) __bf16 v16bf;
typedef __attribute__((ext_vector_type(8)))  float  v8f;

#define M_NBRS   12
#define ATOM_F   64
#define NBR_F    41
#define IN_DIM   169      // 2*64 + 41
#define K_STEPS  6
#define K_PAD    192      // 6 * 32
#define BN_EPS   1e-5f

// workspace layout (float indices into d_ws)
// [0 .. 12288)   : 6*8*512 bf16 W fragments (49152 bytes = 12288 floats)
// [12288..12352) : mean[64]
// [12352..12416) : rstd[64]
#define WSF_MEAN 12288
#define WSF_RSTD 12352

// K index inside a 32-wide k-step for fragment element e on lane `lane`
// (ISA 7.12.2: 16-bit A 16x32 layout; we use the same striping for B)
__device__ __forceinline__ int kmap(int e, int lane) {
    int j = e >> 1, p = e & 1;
    return ((j >= 4) ? 16 : 0) + ((lane >> 4) & 1) * 8 + 2 * (j & 3) + p;
}

// ---------------------------------------------------------------------------
// Kernel 0: repack W (fp32 [128][169]) into bf16 WMMA fragments, K padded to 192.
// frag[(ks*8 + otile)*32 + lane][e] = W[otile*16 + (lane&15)][ks*32 + kmap(e,lane)]
// ---------------------------------------------------------------------------
__global__ void __launch_bounds__(512)
prep_w_kernel(const float* __restrict__ W, __bf16* __restrict__ frag) {
    int bid = blockIdx.x;              // 0..47 = ks*8 + otile
    int ks  = bid >> 3, ot = bid & 7;
    int t   = threadIdx.x;             // 0..511
    int lane = t >> 4, e = t & 15;
    int o = ot * 16 + (lane & 15);
    int k = ks * 32 + kmap(e, lane);
    float v = (k < IN_DIM) ? W[(size_t)o * IN_DIM + k] : 0.0f;
    frag[(size_t)bid * 512 + (size_t)lane * 16 + e] = (__bf16)v;
}

// ---------------------------------------------------------------------------
// Kernel 1: main fused conv. One block = 48 (n,m) rows = 4 whole atoms.
// 384 threads = 12 waves = 3 row-tiles x 4 feature-tiles; each wave computes
// a filter tile (o = ft*16..) and a core tile (o = 64 + ft*16..) via bf16 WMMA,
// applies sigmoid*softplus gate, reduces 12 neighbors per atom, writes x.
// ---------------------------------------------------------------------------
__global__ void __launch_bounds__(384)
cgcnn_gemm_kernel(const float* __restrict__ atom_fea,
                  const float* __restrict__ nbr_fea,
                  const float* __restrict__ bias,
                  const int*   __restrict__ nbr_idx,
                  const __bf16* __restrict__ wfrag,
                  float* __restrict__ x_out) {
    __shared__ __bf16 As[48 * K_PAD];     // 18432 B
    __shared__ float  Msg[48 * ATOM_F];   // 12288 B

    const int t   = threadIdx.x;
    const int blk = blockIdx.x;           // 4 atoms per block

    // ---- stage A tile: 48 rows x 192 bf16 cols into LDS ----
    {
        int row = t >> 3;                 // 0..47
        int seg = t & 7;                  // 8 segments of 24 cols
        long r  = (long)blk * 48 + row;   // global (n,m) row
        int n   = (int)(r / M_NBRS);
        int m   = (int)(r % M_NBRS);
        int nb  = nbr_idx[r];
        const float* selfp = atom_fea + (size_t)n  * ATOM_F;
        const float* nbrp  = atom_fea + (size_t)nb * ATOM_F;
        const float* edgep = nbr_fea + ((size_t)n * M_NBRS + m) * NBR_F;
        int k0 = seg * 24;
        #pragma unroll
        for (int kk = 0; kk < 24; ++kk) {
            int k = k0 + kk;
            float v;
            if      (k < ATOM_F)     v = selfp[k];
            else if (k < 2 * ATOM_F) v = nbrp[k - ATOM_F];
            else if (k < IN_DIM)     v = edgep[k - 2 * ATOM_F];
            else                     v = 0.0f;
            As[row * K_PAD + k] = (__bf16)v;
        }
    }
    __syncthreads();

    // ---- WMMA: 6 k-steps, two accumulators (filter + core) per wave ----
    {
        const int wave    = t >> 5;       // 0..11
        const int lane    = t & 31;
        const int rowtile = wave >> 2;    // 0..2
        const int ft      = wave & 3;     // 0..3 (feature tile)
        const int half    = (lane >> 4) & 1;
        const int col     = lane & 15;    // A row / B col / C col index

        v8f cf = {};
        v8f cc = {};
        const __bf16* arow = As + (size_t)(rowtile * 16 + col) * K_PAD;
        const v16bf*  wf   = (const v16bf*)wfrag;

        #pragma unroll
        for (int ks = 0; ks < K_STEPS; ++ks) {
            v16bf a;
            const __bf16* ap = arow + ks * 32 + half * 8;
            #pragma unroll
            for (int j = 0; j < 8; ++j) {
                int kb = ((j >= 4) ? 16 : 0) + 2 * (j & 3);
                a[2 * j]     = ap[kb];
                a[2 * j + 1] = ap[kb + 1];
            }
            v16bf bfil = wf[(ks * 8 + ft) * 32 + lane];       // W rows [ft*16 .. )
            v16bf bcor = wf[(ks * 8 + 4 + ft) * 32 + lane];   // W rows [64+ft*16 .. )
            cf = __builtin_amdgcn_wmma_f32_16x16x32_bf16(
                     false, a, false, bfil, (short)0, cf, false, false);
            cc = __builtin_amdgcn_wmma_f32_16x16x32_bf16(
                     false, a, false, bcor, (short)0, cc, false, false);
        }

        float bf_ = bias[ft * 16 + col];
        float bc_ = bias[64 + ft * 16 + col];
        #pragma unroll
        for (int v = 0; v < 8; ++v) {
            float fv  = cf[v] + bf_;
            float cv  = cc[v] + bc_;
            float sig = 1.0f / (1.0f + __expf(-fv));
            float sp  = (cv > 20.0f) ? cv : log1pf(__expf(cv));
            int row   = rowtile * 16 + half * 8 + v;
            Msg[row * ATOM_F + ft * 16 + col] = sig * sp;
        }
    }
    __syncthreads();

    // ---- neighbor reduction + residual: 4 atoms x 64 feats = 256 tasks ----
    if (t < 256) {
        int a = t >> 6;
        int f = t & 63;
        float s = 0.0f;
        #pragma unroll
        for (int j = 0; j < M_NBRS; ++j)
            s += Msg[(a * M_NBRS + j) * ATOM_F + f];
        size_t n = (size_t)blk * 4 + a;
        x_out[n * ATOM_F + f] = atom_fea[n * ATOM_F + f] + s;
    }
}

// ---------------------------------------------------------------------------
// Kernel 2: per-feature batch stats (deterministic tree reduce, no atomics).
// ---------------------------------------------------------------------------
__global__ void __launch_bounds__(256)
bn_stats_kernel(const float* __restrict__ x, float* __restrict__ ws, int N) {
    __shared__ float ssum[256], ssq[256];
    int f = blockIdx.x;
    int t = threadIdx.x;
    float s = 0.0f, q = 0.0f;
    for (int n = t; n < N; n += 256) {
        float v = x[(size_t)n * ATOM_F + f];
        s += v; q += v * v;
    }
    ssum[t] = s; ssq[t] = q;
    __syncthreads();
    for (int off = 128; off > 0; off >>= 1) {
        if (t < off) { ssum[t] += ssum[t + off]; ssq[t] += ssq[t + off]; }
        __syncthreads();
    }
    if (t == 0) {
        float mean = ssum[0] / (float)N;
        float var  = ssq[0] / (float)N - mean * mean;
        ws[WSF_MEAN + f] = mean;
        ws[WSF_RSTD + f] = rsqrtf(var + BN_EPS);
    }
}

// ---------------------------------------------------------------------------
// Kernel 3: in-place BatchNorm + softplus on d_out.
// ---------------------------------------------------------------------------
__global__ void __launch_bounds__(256)
bn_softplus_kernel(float* __restrict__ x, const float* __restrict__ ws,
                   const float* __restrict__ gamma, const float* __restrict__ beta,
                   long total) {
    long i = (long)blockIdx.x * blockDim.x + threadIdx.x;
    if (i >= total) return;
    int f = (int)(i & (ATOM_F - 1));
    float v  = x[i];
    float xh = (v - ws[WSF_MEAN + f]) * ws[WSF_RSTD + f];
    float y  = xh * gamma[f] + beta[f];
    x[i] = (y > 20.0f) ? y : log1pf(__expf(y));
}

extern "C" void kernel_launch(void* const* d_in, const int* in_sizes, int n_in,
                              void* d_out, int out_size, void* d_ws, size_t ws_size,
                              hipStream_t stream) {
    const float* atom_fea = (const float*)d_in[0];   // [N,64]
    const float* nbr_fea  = (const float*)d_in[1];   // [N,12,41]
    const float* W        = (const float*)d_in[2];   // [128,169]
    const float* b        = (const float*)d_in[3];   // [128]
    const float* gamma    = (const float*)d_in[4];   // [64]
    const float* beta     = (const float*)d_in[5];   // [64]
    const int*   nbr_idx  = (const int*)d_in[6];     // [N,12]

    __bf16* wfrag = (__bf16*)d_ws;
    float*  wsf   = (float*)d_ws;
    float*  x     = (float*)d_out;

    const int N = in_sizes[0] / ATOM_F;              // 100000

    // 0) repack W into bf16 WMMA fragments (K padded 169 -> 192)
    prep_w_kernel<<<48, 512, 0, stream>>>(W, wfrag);

    // 1) fused gather + GEMM + gate + neighbor-sum + residual
    //    48 (n,m) rows per block == 4 whole atoms => N/4 blocks
    cgcnn_gemm_kernel<<<N / 4, 384, 0, stream>>>(atom_fea, nbr_fea, b, nbr_idx,
                                                 wfrag, x);

    // 2) batch-norm statistics (one block per feature)
    bn_stats_kernel<<<ATOM_F, 256, 0, stream>>>(x, wsf, N);

    // 3) normalize + affine + softplus, in place on d_out
    long total = (long)N * ATOM_F;
    bn_softplus_kernel<<<(int)((total + 255) / 256), 256, 0, stream>>>(
        x, wsf, gamma, beta, total);
}
